// RNN_19172734009489
// MI455X (gfx1250) — matrix operand order repaired and checked by reading, not compiled, not run on hardware
//
#include <hip/hip_runtime.h>

// ---------------------------------------------------------------------------
// Types for CDNA5 WMMA (wave32): v_wmma_f32_16x16x32_bf16
// ---------------------------------------------------------------------------
typedef __attribute__((ext_vector_type(16))) __bf16 v16bf;
typedef __attribute__((ext_vector_type(8)))  float  v8f;
typedef __attribute__((ext_vector_type(4)))  int    v4i;

// Pointer types with the pointee in explicit address spaces (so the
// attribute actually binds), for the async global->LDS builtin.
typedef __attribute__((address_space(1))) v4i glob_v4i;
typedef __attribute__((address_space(3))) v4i lds_v4i;

union FragBF {
    v16bf v;
    uint4 q[2];
};

// CDNA5 async global->LDS path (ASYNCcnt), if the toolchain exposes it.
#if __has_builtin(__builtin_amdgcn_global_load_async_to_lds_b128) && \
    __has_builtin(__builtin_amdgcn_s_wait_asynccnt)
#define USE_ASYNC_LDS 1
#else
#define USE_ASYNC_LDS 0
#endif

__device__ __forceinline__ void async_fence() {
#if USE_ASYNC_LDS
    __builtin_amdgcn_s_wait_asynccnt(0);
#endif
}

// Copy 16 bytes global -> LDS (async when available).
__device__ __forceinline__ void stage16(const unsigned short* __restrict__ gsrc,
                                        unsigned short* lds_dst) {
#if USE_ASYNC_LDS
    __builtin_amdgcn_global_load_async_to_lds_b128(
        (glob_v4i*)gsrc, (lds_v4i*)lds_dst, 0, 0);
#else
    *(uint4*)lds_dst = *(const uint4*)gsrc;
#endif
}

__device__ __forceinline__ unsigned short f2bf(float f) {
    union { float f; unsigned int u; } cv;
    cv.f = f;
    unsigned int u = cv.u;
    u += 0x7fffu + ((u >> 16) & 1u);   // round-to-nearest-even
    return (unsigned short)(u >> 16);
}

// Problem constants
#define BATCH   8
#define SEQ     4096
#define DIN     512
#define DHID    1024
#define DOUT    512
#define MROWS   (BATCH * SEQ)     // 32768
#define CHUNK   256
#define NCHUNK  (SEQ / CHUNK)     // 16

// GEMM tiling
#define TM 128
#define TN 128
#define TK 32

// ---------------------------------------------------------------------------
// Gate precompute: a = sigmoid(a_logit); mul = sqrt(1-a^2); aL = a^CHUNK
// ---------------------------------------------------------------------------
__global__ void prep_gates(const float* __restrict__ alog,
                           float* __restrict__ a_g,
                           float* __restrict__ mul_g,
                           float* __restrict__ aL_g) {
    int h = blockIdx.x * blockDim.x + threadIdx.x;
    if (h >= DHID) return;
    float l  = alog[h];
    float av = 1.0f / (1.0f + __expf(-l));
    a_g[h]   = av;
    mul_g[h] = sqrtf(fmaxf(1.0f - av * av, 0.0f));
    float p = av;
#pragma unroll
    for (int i = 0; i < 8; ++i) p *= p;   // a^256
    aL_g[h] = p;
}

// ---------------------------------------------------------------------------
// f32 -> bf16 bulk convert (n4 = elements/4)
// ---------------------------------------------------------------------------
__global__ void conv_bf16_x4(const float* __restrict__ src,
                             unsigned short* __restrict__ dst, int n4) {
    int i = blockIdx.x * blockDim.x + threadIdx.x;
    if (i >= n4) return;
    float4 f = ((const float4*)src)[i];
    uint2 o;
    o.x = (unsigned)f2bf(f.x) | ((unsigned)f2bf(f.y) << 16);
    o.y = (unsigned)f2bf(f.z) | ((unsigned)f2bf(f.w) << 16);
    ((uint2*)dst)[i] = o;
}

// Transposed bf16 W_in: dst[n][k] = W_in[k][n]  (n<1024, k<512)
__global__ void build_winT(const float* __restrict__ Win,
                           unsigned short* __restrict__ dst) {
    int i = blockIdx.x * blockDim.x + threadIdx.x;   // < 1024*512
    if (i >= DHID * DIN) return;
    int n = i >> 9;          // /512
    int k = i & 511;
    dst[i] = f2bf(Win[(size_t)k * DHID + n]);
}

// Transposed concatenated bf16 weight: dst[n][k] = [W_dx;W_out][k][n]
// (n<512, k<1536)
__global__ void build_wcat2T(const float* __restrict__ Wdx,
                             const float* __restrict__ Wout,
                             unsigned short* __restrict__ dst) {
    int i = blockIdx.x * blockDim.x + threadIdx.x;   // < 512*1536
    if (i >= DOUT * (DIN + DHID)) return;
    int n = i / (DIN + DHID);
    int k = i - n * (DIN + DHID);
    float f = (k < DIN) ? Wdx[(size_t)k * DOUT + n]
                        : Wout[(size_t)(k - DIN) * DOUT + n];
    dst[i] = f2bf(f);
}

// ---------------------------------------------------------------------------
// WMMA GEMM, 256 threads = 8 waves, block tile 128x128, K-step 32.
// B is pre-transposed [N][K], so A and B staging are identical b128 copies.
// MODE 0: u = (x @ W_in + b_in) * mul              (A = x_bf16, K=512,  N=1024)
// MODE 1: out = 0.5*([x|h] @ Wcat2 + b_dx + b_out) (K=1536, N=512)
// ---------------------------------------------------------------------------
template <int MODE>
__global__ __launch_bounds__(256)
void gemm_wmma(const unsigned short* __restrict__ Ax,    // x_bf16 (lda=512)
               const unsigned short* __restrict__ Ah,    // h_bf16 (lda=1024), MODE 1
               const unsigned short* __restrict__ Bt,    // weight bf16 [N][K]
               const float* __restrict__ bias0,
               const float* __restrict__ bias1,          // MODE 1 only
               const float* __restrict__ mul_g,          // MODE 0 only
               float* __restrict__ Out,
               int N, int K) {
    __shared__ __align__(16) unsigned short Asm[2][TM * TK];
    __shared__ __align__(16) unsigned short Bsm[2][TN * TK];

    const int tid  = threadIdx.x;
    const int lane = tid & 31;
    const int wave = tid >> 5;
    const int wm   = (wave >> 1) * 32;   // wave M offset: 0,32,64,96
    const int wn   = (wave & 1) * 64;    // wave N offset: 0,64
    const int r    = lane & 15;
    const int hi   = lane >> 4;

    const int NK = K / TK;
    const int row = tid >> 1;            // 0..127 (shared by A and B staging)
    const int kh  = (tid & 1) * 16;      // 0 or 16

    auto loadTiles = [&](int ks, int buf) {
        const int kk0 = ks * TK;
        // ---- A tile: 128 rows x 32 K, row-major in LDS
        {
            int grow = blockIdx.x * TM + row;
            const unsigned short* src;
            if (MODE == 0) {
                src = Ax + (size_t)grow * DIN + kk0 + kh;
            } else {
                src = (kk0 < DIN) ? (Ax + (size_t)grow * DIN + kk0 + kh)
                                  : (Ah + (size_t)grow * DHID + (kk0 - DIN) + kh);
            }
            stage16(src,     &Asm[buf][row * TK + kh]);
            stage16(src + 8, &Asm[buf][row * TK + kh + 8]);
        }
        // ---- B tile: 128 n-rows x 32 K from pre-transposed weights
        {
            const unsigned short* src =
                Bt + (size_t)(blockIdx.y * TN + row) * K + kk0 + kh;
            stage16(src,     &Bsm[buf][row * TK + kh]);
            stage16(src + 8, &Bsm[buf][row * TK + kh + 8]);
        }
    };

    v8f acc[2][4];
#pragma unroll
    for (int mi = 0; mi < 2; ++mi)
#pragma unroll
        for (int ni = 0; ni < 4; ++ni)
#pragma unroll
            for (int e = 0; e < 8; ++e) acc[mi][ni][e] = 0.0f;

    loadTiles(0, 0);
    async_fence();
    __syncthreads();

    for (int ks = 0; ks < NK; ++ks) {
        const int cur = ks & 1;
        if (ks + 1 < NK) loadTiles(ks + 1, 1 - cur);

        // prefetch the k+2 A tile (global_prefetch_b8)
        if (ks + 2 < NK) {
            int kk2  = (ks + 2) * TK;
            int grow = blockIdx.x * TM + row;
            const unsigned short* pf;
            if (MODE == 0 || kk2 < DIN)
                pf = Ax + (size_t)grow * DIN + (kk2 < DIN ? kk2 : 0);
            else
                pf = Ah + (size_t)grow * DHID + (kk2 - DIN);
            __builtin_prefetch(pf, 0, 0);
        }

        // A fragments: lane(r,hi) holds K {8hi..8hi+7} ++ {16+8hi..16+8hi+7}
        FragBF af[2];
#pragma unroll
        for (int mi = 0; mi < 2; ++mi) {
            int base = (wm + mi * 16 + r) * TK + hi * 8;
            af[mi].q[0] = *(const uint4*)&Asm[cur][base];
            af[mi].q[1] = *(const uint4*)&Asm[cur][base + 16];
        }
        // B fragments: lane(c,hi) holds K {8hi..8hi+7} ++ {16+8hi..16+8hi+7}
        FragBF bf[4];
#pragma unroll
        for (int ni = 0; ni < 4; ++ni) {
            int base = (wn + ni * 16 + r) * TK + hi * 8;
            bf[ni].q[0] = *(const uint4*)&Bsm[cur][base];
            bf[ni].q[1] = *(const uint4*)&Bsm[cur][base + 16];
        }

#pragma unroll
        for (int mi = 0; mi < 2; ++mi)
#pragma unroll
            for (int ni = 0; ni < 4; ++ni)
                acc[mi][ni] = __builtin_amdgcn_wmma_f32_16x16x32_bf16(
                    false, af[mi].v, false, bf[ni].v,
                    (short)0, acc[mi][ni], false, false);

        async_fence();
        __syncthreads();
    }

    // Epilogue: C element v -> row = v + 8*hi, col = lane&15
#pragma unroll
    for (int mi = 0; mi < 2; ++mi) {
#pragma unroll
        for (int ni = 0; ni < 4; ++ni) {
            int col = blockIdx.y * TN + wn + ni * 16 + r;
            float b0 = bias0[col];
            float b1 = (MODE == 1) ? bias1[col] : 0.0f;
            float mg = (MODE == 0) ? mul_g[col] : 0.0f;
#pragma unroll
            for (int v = 0; v < 8; ++v) {
                int orow = blockIdx.x * TM + wm + mi * 16 + v + 8 * hi;
                float val = acc[mi][ni][v];
                if (MODE == 0)
                    val = (val + b0) * mg;
                else
                    val = 0.5f * (val + b0 + b1);
                Out[(size_t)orow * N + col] = val;
            }
        }
    }
}

// ---------------------------------------------------------------------------
// Chunked linear scan: h_t = a*h_{t-1} + u_t
// Pass A: local scans per chunk (in-place over U), record chunk-final values.
// ---------------------------------------------------------------------------
__global__ void scanA(float* __restrict__ U, float* __restrict__ last,
                      const float* __restrict__ a_g) {
    int h = blockIdx.x * blockDim.x + threadIdx.x;   // 0..1023
    int c = blockIdx.y;                              // chunk
    int b = blockIdx.z;                              // batch
    float av = a_g[h];
    size_t base = ((size_t)b * SEQ + (size_t)c * CHUNK) * DHID + h;
    float hl = 0.0f;
    for (int t = 0; t < CHUNK; ++t) {
        hl = av * hl + U[base + (size_t)t * DHID];
        U[base + (size_t)t * DHID] = hl;
    }
    last[((size_t)b * NCHUNK + c) * DHID + h] = hl;
}

// Pass B: sequential combine across 16 chunks; emits carry-in per chunk,
// seeds with h0, and writes h_last directly to the output tail.
__global__ void scanB(const float* __restrict__ last,
                      float* __restrict__ carry,
                      const float* __restrict__ aL_g,
                      const float* __restrict__ h0,
                      float* __restrict__ hlast_out) {
    int idx = blockIdx.x * blockDim.x + threadIdx.x;   // 0..8191
    int b = idx >> 10;
    int h = idx & (DHID - 1);
    float al   = aL_g[h];
    float full = h0[idx];
    for (int c = 0; c < NCHUNK; ++c) {
        size_t o = ((size_t)b * NCHUNK + c) * DHID + h;
        carry[o] = full;
        full = al * full + last[o];
    }
    hlast_out[idx] = full;
}

// Pass C: h_full[t] = h_local[t] + a^{t+1} * carry; emit bf16 copy for GEMM2.
__global__ void scanC(const float* __restrict__ Hl,
                      const float* __restrict__ carry,
                      const float* __restrict__ a_g,
                      unsigned short* __restrict__ Hbf) {
    int h = blockIdx.x * blockDim.x + threadIdx.x;
    int c = blockIdx.y;
    int b = blockIdx.z;
    float av = a_g[h];
    float cr = carry[((size_t)b * NCHUNK + c) * DHID + h];
    size_t base = ((size_t)b * SEQ + (size_t)c * CHUNK) * DHID + h;
    float p = av;
    for (int t = 0; t < CHUNK; ++t) {
        float val = Hl[base + (size_t)t * DHID] + p * cr;
        p *= av;
        Hbf[base + (size_t)t * DHID] = f2bf(val);
    }
}

// ---------------------------------------------------------------------------
// kernel_launch
// Inputs: 0:x 1:h0 2:a_logit 3:W_dx 4:b_dx 5:W_in 6:b_in 7:W_out 8:b_out
// Output: out (8*4096*512 f32) ++ h_last (8*1024 f32)
// ---------------------------------------------------------------------------
extern "C" void kernel_launch(void* const* d_in, const int* in_sizes, int n_in,
                              void* d_out, int out_size, void* d_ws, size_t ws_size,
                              hipStream_t stream) {
    const float* x      = (const float*)d_in[0];
    const float* h0     = (const float*)d_in[1];
    const float* a_log  = (const float*)d_in[2];
    const float* W_dx   = (const float*)d_in[3];
    const float* b_dx   = (const float*)d_in[4];
    const float* W_in   = (const float*)d_in[5];
    const float* b_in   = (const float*)d_in[6];
    const float* W_out  = (const float*)d_in[7];
    const float* b_out  = (const float*)d_in[8];

    float* out_f   = (float*)d_out;
    float* hlast_f = out_f + (size_t)MROWS * DOUT;

    // Workspace carve-up
    char* ws = (char*)d_ws;
    size_t o = 0;
    unsigned short* xbf    = (unsigned short*)(ws + o); o += (size_t)MROWS * DIN * 2;        // 32 MB
    unsigned short* winT   = (unsigned short*)(ws + o); o += (size_t)DHID * DIN * 2;         // 1 MB
    unsigned short* wcat2T = (unsigned short*)(ws + o); o += (size_t)DOUT * (DIN + DHID) * 2;// 1.5 MB
    float*          ubuf   = (float*)(ws + o);          o += (size_t)MROWS * DHID * 4;       // 128 MB
    unsigned short* hbf    = (unsigned short*)(ws + o); o += (size_t)MROWS * DHID * 2;       // 64 MB
    float*          lastb  = (float*)(ws + o);          o += (size_t)BATCH * NCHUNK * DHID * 4;
    float*          carry  = (float*)(ws + o);          o += (size_t)BATCH * NCHUNK * DHID * 4;
    float*          a_g    = (float*)(ws + o);          o += DHID * 4;
    float*          mul_g  = (float*)(ws + o);          o += DHID * 4;
    float*          aL_g   = (float*)(ws + o);          o += DHID * 4;
    (void)ws_size; (void)in_sizes; (void)n_in; (void)out_size;

    // 1) gates
    prep_gates<<<dim3(4), dim3(256), 0, stream>>>(a_log, a_g, mul_g, aL_g);

    // 2) bf16 conversions (+ weight transposes)
    {
        int n4 = (MROWS * DIN) / 4;                    // x
        conv_bf16_x4<<<dim3((n4 + 255) / 256), dim3(256), 0, stream>>>(x, xbf, n4);
    }
    build_winT<<<dim3((DHID * DIN) / 256), dim3(256), 0, stream>>>(W_in, winT);
    build_wcat2T<<<dim3((DOUT * (DIN + DHID)) / 256), dim3(256), 0, stream>>>(W_dx, W_out, wcat2T);

    // 3) GEMM1: u = (x @ W_in + b_in) * sqrt(1-a^2)
    gemm_wmma<0><<<dim3(MROWS / TM, DHID / TN), dim3(256), 0, stream>>>(
        xbf, nullptr, winT, b_in, nullptr, mul_g, ubuf, DHID, DIN);

    // 4) chunked scan
    scanA<<<dim3(DHID / 256, NCHUNK, BATCH), dim3(256), 0, stream>>>(ubuf, lastb, a_g);
    scanB<<<dim3((BATCH * DHID) / 256), dim3(256), 0, stream>>>(lastb, carry, aL_g, h0, hlast_f);
    scanC<<<dim3(DHID / 256, NCHUNK, BATCH), dim3(256), 0, stream>>>(ubuf, carry, a_g, hbf);

    // 5) GEMM2: out = 0.5*([x|h] @ [W_dx;W_out] + b_dx + b_out)
    gemm_wmma<1><<<dim3(MROWS / TM, DOUT / TN), dim3(256), 0, stream>>>(
        xbf, hbf, wcat2T, b_dx, b_out, nullptr, out_f, DOUT, DIN + DHID);
}
